// InfoNCE_71665824301703
// MI455X (gfx1250) — matrix operand order repaired and checked by reading, not compiled, not run on hardware
//
#include <hip/hip_runtime.h>
#include <hip/hip_fp16.h>

typedef __attribute__((ext_vector_type(16))) _Float16 v16h;
typedef __attribute__((ext_vector_type(8)))  float    v8f;

#define N_TOT 8192   // 2*bs
#define BS    4096
#define D     512
#define INV_T 10.0f  // 1/TEMPERATURE
#define ROWB  144    // LDS bytes per staged row: 128B data + 16B pad (bank-perfect)

union Frag16 {
  v16h v;
  uint4 q[2];
};

// Async global->LDS copy (CDNA5, tracked by ASYNCcnt). VDST = LDS byte offset.
__device__ __forceinline__ void async_copy_b128(unsigned lds_off, const char* gptr) {
  asm volatile("global_load_async_to_lds_b128 %0, %1, off"
               :: "v"(lds_off), "v"(gptr)
               : "memory");
}
__device__ __forceinline__ void wait_async0() {
  asm volatile("s_wait_asynccnt 0" ::: "memory");
}
// Low 32 bits of a flat LDS pointer are the LDS byte offset (ISA 10.2 aperture map).
__device__ __forceinline__ unsigned lds_off32(const void* p) {
  return (unsigned)(unsigned long long)p;
}

// ---------------------------------------------------------------------------
// Zero the accumulators (rowsum[N_TOT] + possum scalar) in workspace.
// ---------------------------------------------------------------------------
__global__ void zero_kernel(float* __restrict__ p, int n) {
  int i = blockIdx.x * blockDim.x + threadIdx.x;
  if (i < n) p[i] = 0.0f;
}

// ---------------------------------------------------------------------------
// Row-normalize reps = concat(z_i, z_j) and quantize to f16.
// One wave per row; each lane handles 16 consecutive floats.
// ---------------------------------------------------------------------------
__global__ __launch_bounds__(256) void norm_kernel(const float* __restrict__ zi,
                                                   const float* __restrict__ zj,
                                                   _Float16* __restrict__ R) {
  const int wave = threadIdx.x >> 5;
  const int lane = threadIdx.x & 31;
  const int row  = blockIdx.x * 8 + wave;

  const float* src = (row < BS) ? (zi + (size_t)row * D)
                                : (zj + (size_t)(row - BS) * D);
  const float4* s4 = (const float4*)src + lane * 4;

  float4 f[4];
  float ss = 0.0f;
#pragma unroll
  for (int k = 0; k < 4; ++k) {
    f[k] = s4[k];
    ss += f[k].x * f[k].x + f[k].y * f[k].y + f[k].z * f[k].z + f[k].w * f[k].w;
  }
#pragma unroll
  for (int m = 1; m < 32; m <<= 1) ss += __shfl_xor(ss, m, 32);

  const float inv = 1.0f / fmaxf(sqrtf(ss), 1e-8f);

  union { _Float16 h[16]; uint4 q[2]; } o;
#pragma unroll
  for (int k = 0; k < 4; ++k) {
    o.h[4 * k + 0] = (_Float16)(f[k].x * inv);
    o.h[4 * k + 1] = (_Float16)(f[k].y * inv);
    o.h[4 * k + 2] = (_Float16)(f[k].z * inv);
    o.h[4 * k + 3] = (_Float16)(f[k].w * inv);
  }
  uint4* d4 = (uint4*)(R + (size_t)row * D) + lane * 2;
  d4[0] = o.q[0];
  d4[1] = o.q[1];
}

// ---------------------------------------------------------------------------
// sim = R * R^T via v_wmma_f32_16x16x32_f16 with LDS double-buffered staging
// through global_load_async_to_lds_b128, fused exp + row reduction.
// Block = 256 threads = 8 waves (2 M x 4 N); wave tile 32x64, block tile
// 64 x 256, k staged in 64-wide slabs (8 stages, 2 WMMA k-substeps each).
// Grid = (8192/64, 8192/256).
// ---------------------------------------------------------------------------
__global__ __launch_bounds__(256) void simexp_kernel(const _Float16* __restrict__ R,
                                                     float* __restrict__ rowsum,
                                                     float* __restrict__ possum) {
  __shared__ __align__(16) char sA[2][64 * ROWB];    // block A rows, 64B*2 per row
  __shared__ __align__(16) char sB[2][256 * ROWB];   // block B rows
  __shared__ float lrow[64];
  __shared__ float lpos;

  const int tid = threadIdx.x;
  if (tid < 64) lrow[tid] = 0.0f;
  if (tid == 0) lpos = 0.0f;

  const int wave = tid >> 5;
  const int lane = tid & 31;
  const int lm   = lane & 15;
  const int lh   = lane >> 4;
  const int mw   = wave & 1;
  const int nw   = wave >> 1;

  const int M0 = blockIdx.x * 64 + mw * 32;
  const int N0 = blockIdx.y * 256 + nw * 64;

  // --- async staging bases: chunk c = tid + 256*j -> sub = tid&7 (fixed),
  // row r = (tid>>3) + 32*j. Rows 0..63 = A tile, 64..319 = B tile.
  const char* Rb = (const char*)R;
  const int sub = tid & 7;          // 16B sub-chunk within the 128B row slab
  const int r0  = tid >> 3;         // 0..31
  const char* gA = Rb + ((size_t)(blockIdx.x * 64 + r0) * D) * 2 + sub * 16;
  const char* gB = Rb + ((size_t)(blockIdx.y * 256 + r0) * D) * 2 + sub * 16;
  unsigned loffA[2], loffB[2];
#pragma unroll
  for (int b = 0; b < 2; ++b) {
    loffA[b] = lds_off32(&sA[b][r0 * ROWB + sub * 16]);
    loffB[b] = lds_off32(&sB[b][r0 * ROWB + sub * 16]);
  }

  auto issue = [&](int buf, int kbyte) {
    const char* g = gA + kbyte;
    unsigned l = loffA[buf];
#pragma unroll
    for (int j = 0; j < 2; ++j) {               // 64 A rows
      async_copy_b128(l, g);
      g += 32 * (D * 2);
      l += 32 * ROWB;
    }
    g = gB + kbyte;
    l = loffB[buf];
#pragma unroll
    for (int j = 0; j < 8; ++j) {               // 256 B rows
      async_copy_b128(l, g);
      g += 32 * (D * 2);
      l += 32 * ROWB;
    }
  };

  v8f acc[2][4] = {};

  issue(0, 0);
#pragma unroll
  for (int st = 0; st < 8; ++st) {
    const int cur = st & 1;
    const int nxt = cur ^ 1;
    wait_async0();          // my stage-`st` copies have landed in LDS
    __syncthreads();        // everyone's landed; everyone done reading buf `nxt`
    if (st < 7) issue(nxt, (st + 1) * 128);   // overlap next stage with WMMAs

#pragma unroll
    for (int s = 0; s < 2; ++s) {             // two 32-wide k-substeps
      Frag16 a[2];
#pragma unroll
      for (int mt = 0; mt < 2; ++mt) {
        // A frag (16x32): lane lm = row; low lanes K{0..7,16..23}, high {8..15,24..31}
        const char* ap = &sA[cur][(mw * 32 + 16 * mt + lm) * ROWB + 64 * s];
        a[mt].q[0] = *(const uint4*)(ap + 16 * lh);
        a[mt].q[1] = *(const uint4*)(ap + 32 + 16 * lh);
      }
#pragma unroll
      for (int t = 0; t < 4; ++t) {
        // B frag (32x16): lane lm = column (= R row); low lanes K{0..7,16..23}, high {8..15,24..31}
        const char* bp = &sB[cur][(nw * 64 + 16 * t + lm) * ROWB + 64 * s + 16 * lh];
        Frag16 b;
        b.q[0] = ((const uint4*)bp)[0];
        b.q[1] = ((const uint4*)bp)[2];
        acc[0][t] = __builtin_amdgcn_wmma_f32_16x16x32_f16(
            false, a[0].v, false, b.v, (short)0, acc[0][t], false, false);
        acc[1][t] = __builtin_amdgcn_wmma_f32_16x16x32_f16(
            false, a[1].v, false, b.v, (short)0, acc[1][t], false, false);
      }
    }
  }

  // Epilogue: C/D layout — lane lm = column; VGPR v = row M0 + 16*mt + v + 8*lh.
  float pp = 0.0f;
#pragma unroll
  for (int mt = 0; mt < 2; ++mt) {
    float rp[8] = {0.f, 0.f, 0.f, 0.f, 0.f, 0.f, 0.f, 0.f};
#pragma unroll
    for (int t = 0; t < 4; ++t) {
      const int col = N0 + 16 * t + lm;
#pragma unroll
      for (int v = 0; v < 8; ++v) {
        const int rowg = M0 + 16 * mt + v + 8 * lh;
        const float s = acc[mt][t][v];
        const float e = (col == rowg) ? 0.0f : __expf(s * INV_T);  // mask diagonal
        rp[v] += e;
        if (col == (rowg ^ BS)) pp += s;    // positive pair (row+bs mod 2bs)
      }
    }
#pragma unroll
    for (int v = 0; v < 8; ++v) {
      float r = rp[v];
      r += __shfl_xor(r, 1, 32);
      r += __shfl_xor(r, 2, 32);
      r += __shfl_xor(r, 4, 32);
      r += __shfl_xor(r, 8, 32);
      if (lm == 0) atomicAdd(&lrow[mw * 32 + 16 * mt + v + 8 * lh], r);  // ds_add_f32
    }
  }
  pp += __shfl_xor(pp, 1, 32);
  pp += __shfl_xor(pp, 2, 32);
  pp += __shfl_xor(pp, 4, 32);
  pp += __shfl_xor(pp, 8, 32);
  pp += __shfl_xor(pp, 16, 32);
  if (lane == 0) atomicAdd(&lpos, pp);

  __syncthreads();
  if (tid < 64) atomicAdd(&rowsum[blockIdx.x * 64 + tid], lrow[tid]);
  if (tid == 0) atomicAdd(possum, lpos);
}

// ---------------------------------------------------------------------------
// loss = ( sum_i log(denom_i) - (1/T) * sum_i pos_i ) / N
// ---------------------------------------------------------------------------
__global__ __launch_bounds__(256) void finalize_kernel(const float* __restrict__ rowsum,
                                                       const float* __restrict__ possum,
                                                       float* __restrict__ out) {
  __shared__ float red[256];
  float a = 0.0f;
  for (int i = threadIdx.x; i < N_TOT; i += 256) a += __logf(rowsum[i]);
  red[threadIdx.x] = a;
  __syncthreads();
  for (int s = 128; s > 0; s >>= 1) {
    if (threadIdx.x < s) red[threadIdx.x] += red[threadIdx.x + s];
    __syncthreads();
  }
  if (threadIdx.x == 0)
    out[0] = (red[0] - possum[0] * INV_T) * (1.0f / (float)N_TOT);
}

// ---------------------------------------------------------------------------
// Host launcher
// ---------------------------------------------------------------------------
extern "C" void kernel_launch(void* const* d_in, const int* in_sizes, int n_in,
                              void* d_out, int out_size, void* d_ws, size_t ws_size,
                              hipStream_t stream) {
  const float* zi = (const float*)d_in[0];  // [4096, 512] f32
  const float* zj = (const float*)d_in[1];  // [4096, 512] f32
  float* out = (float*)d_out;               // scalar f32

  // Workspace: R f16 [8192*512] (8 MB) | rowsum f32[8192] | possum f32[1]
  _Float16* R   = (_Float16*)d_ws;
  float* rowsum = (float*)((char*)d_ws + (size_t)N_TOT * D * sizeof(_Float16));
  float* possum = rowsum + N_TOT;

  zero_kernel<<<(N_TOT + 1 + 255) / 256, 256, 0, stream>>>(rowsum, N_TOT + 1);
  norm_kernel<<<N_TOT / 8, 256, 0, stream>>>(zi, zj, R);
  simexp_kernel<<<dim3(N_TOT / 64, N_TOT / 256), 256, 0, stream>>>(R, rowsum, possum);
  finalize_kernel<<<1, 256, 0, stream>>>(rowsum, possum, out);
}
